// xDeepFM_26680336842928
// MI455X (gfx1250) — matrix operand order repaired and checked by reading, not compiled, not run on hardware
//
#include <hip/hip_runtime.h>
#include <hip/hip_bf16.h>

// xDeepFM forward for MI455X (gfx1250, wave32).
// Dominant cost: CIN einsums (~34 GFLOP) -> bf16 WMMA GEMMs with fp32
// accumulate (v_wmma_f32_16x16x32_bf16). Outer-product A-tiles built in LDS
// (layer1 tile = 16x3328 bf16 = 104KB of the 320KB WGP LDS). Weights are
// pre-swizzled into WMMA B-fragment order so the inner loop is pure
// ds_load_b128 + global_load_b128 + wmma. Dual accumulator chains break the
// WMMA RAW dependence and deepen the load pipeline.

#define BATCH   4096
#define NDENSE  13
#define NSPARSE 26
#define VOCAB   100000
#define EMBD    8
#define H0W     224            // 221 padded (zeros) for the K dimension
#define KROWS   (BATCH * 8)    // (b,k) rows for the CIN GEMMs

typedef __attribute__((ext_vector_type(16))) __bf16 v16bf;
typedef __attribute__((ext_vector_type(8)))  __bf16 v8bf;
typedef __attribute__((ext_vector_type(8)))  float  v8f;

union V16 { v16bf v; v8bf h[2]; };

// ---------------------------------------------------------------------------
// Kernel 1: embedding gather + MLP input assembly + linear head
// ---------------------------------------------------------------------------
__global__ void prep_gather(const float* __restrict__ inputs,
                            const float* __restrict__ emb,
                            const float* __restrict__ linW,
                            const float* __restrict__ linb,
                            float*  __restrict__ X0f,     // (B,26,8) f32
                            __bf16* __restrict__ h0,      // (B,224) bf16
                            float*  __restrict__ linout)  // (B,)
{
  const int b = blockIdx.x;
  const int t = threadIdx.x;
  const float* row = inputs + (size_t)b * 39;

  if (t < NDENSE) {
    h0[(size_t)b * H0W + t] = (__bf16)row[t];
  } else if (t < NDENSE + NSPARSE * EMBD) {
    const int u = t - NDENSE;
    const int s = u >> 3, e = u & 7;
    const int idx = (int)row[NDENSE + s];
    const float v = emb[((size_t)s * VOCAB + idx) * EMBD + e];
    X0f[((size_t)b * NSPARSE + s) * EMBD + e] = v;
    h0[(size_t)b * H0W + t] = (__bf16)v;
  } else if (t < H0W) {
    h0[(size_t)b * H0W + t] = (__bf16)0.f;   // K padding
  } else if (t == 255) {
    float acc = linb[0];
    #pragma unroll
    for (int c = 0; c < 39; ++c) acc += row[c] * linW[c];
    linout[b] = acc;
  }
}

// ---------------------------------------------------------------------------
// Kernel 2: weight -> bf16 WMMA B-fragment swizzle.
// B frag for 16x16x32: lane l holds N-col (l%16); elements stripe K (table order).
// Each lane's 16 bf16 are stored contiguously (one 32B load per fragment).
// ---------------------------------------------------------------------------
__global__ void swizzle_wfrag(const float* __restrict__ src,
                              __bf16* __restrict__ dst,
                              int Kfeat, int N, int nMajor, int total)
{
  const int id = blockIdx.x * blockDim.x + threadIdx.x;
  if (id >= total) return;
  const int NT = N >> 4;
  const int perKt = NT * 512;
  const int kt   = id / perKt;
  const int rem  = id - kt * perKt;
  const int nt   = rem >> 9;
  const int r    = rem & 511;
  const int lane = r >> 4;
  const int e    = r & 15;
  const int k = kt * 32 + ((lane >> 4) << 3) + (e & 7) + ((e >> 3) << 4);
  const int n = nt * 16 + (lane & 15);
  float v = 0.f;
  if (k < Kfeat) v = nMajor ? src[(size_t)n * Kfeat + k]    // W_cin: (o=N, f=K)
                            : src[(size_t)k * N + n];       // d_W*:  (K, N)
  dst[id] = (__bf16)v;
}

// ---------------------------------------------------------------------------
// Fragment helpers
// ---------------------------------------------------------------------------
__device__ __forceinline__ v16bf load_afrag(const __bf16* arow) {
  V16 a;
  a.h[0] = *(const v8bf*)(arow);        // k 0-7  (lanes<16) / 8-15
  a.h[1] = *(const v8bf*)(arow + 16);   // k 16-23           / 24-31
  return a.v;
}

// ---------------------------------------------------------------------------
// Kernel 3: CIN layer. Row r = b*8+k. A_row = outer(Xi[b,:,k], X0[b,:,k]),
// built as bf16 in LDS (16 rows x KPAD). 8 waves; wave w owns n-tile w of the
// 128 output channels. Dual acc chains over even/odd kt.
// ---------------------------------------------------------------------------
template<int KPAD, int KFEAT, int IDIM>
__global__ void cin_layer(const float* __restrict__ Xi,    // (B,IDIM,8)
                          const float* __restrict__ X0f,   // (B,26,8)
                          const __bf16* __restrict__ Wfrag,
                          const float* __restrict__ bias,  // (128,)
                          float* __restrict__ outp)        // (B,128,8)
{
  extern __shared__ __bf16 lA[];                           // 16 * KPAD
  const int r0 = blockIdx.x * 16;

  for (int t = threadIdx.x; t < 16 * KPAD; t += 256) {
    const int rowi = t / KPAD;
    const int f = t - rowi * KPAD;
    const int rr = r0 + rowi;
    const int b = rr >> 3, k = rr & 7;
    float v = 0.f;
    if (f < KFEAT) {
      const int i = f / 26, j = f - i * 26;
      v = Xi[((size_t)b * IDIM + i) * 8 + k] * X0f[((size_t)b * 26 + j) * 8 + k];
    }
    lA[rowi * KPAD + f] = (__bf16)v;
  }
  __syncthreads();

  const int wave = threadIdx.x >> 5;     // n-tile 0..7
  const int lane = threadIdx.x & 31;
  const int m  = lane & 15;
  const int ka = (lane < 16) ? 0 : 8;    // 16-bit A layout half-wave K split
  const __bf16* arow = lA + m * KPAD + ka;
  const __bf16* bfr  = Wfrag + ((size_t)wave * 32 + lane) * 16;

  constexpr int KT = KPAD / 32;
  static_assert((KT & 1) == 0, "CIN KT must be even");
  v8f acc0 = {0.f, 0.f, 0.f, 0.f, 0.f, 0.f, 0.f, 0.f};
  v8f acc1 = acc0;
  #pragma unroll 2
  for (int kt = 0; kt < KT; kt += 2) {
    const v16bf a0 = load_afrag(arow + (kt    ) * 32);
    const v16bf b0 = *(const v16bf*)(bfr + (size_t)(kt    ) * (8 * 512));
    const v16bf a1 = load_afrag(arow + (kt + 1) * 32);
    const v16bf b1 = *(const v16bf*)(bfr + (size_t)(kt + 1) * (8 * 512));
    acc0 = __builtin_amdgcn_wmma_f32_16x16x32_bf16(false, a0, false, b0,
                                                   (short)0, acc0, false, false);
    acc1 = __builtin_amdgcn_wmma_f32_16x16x32_bf16(false, a1, false, b1,
                                                   (short)0, acc1, false, false);
  }

  const int o  = wave * 16 + (lane & 15);
  const int mb = (lane < 16) ? 0 : 8;    // C layout: lanes>=16 hold M=8..15
  const float bv = bias[o];
  #pragma unroll
  for (int v = 0; v < 8; ++v) {
    const int rr = r0 + mb + v;
    const int b = rr >> 3, k = rr & 7;
    outp[((size_t)b * 128 + o) * 8 + k] = acc0[v] + acc1[v] + bv;
  }
}

// ---------------------------------------------------------------------------
// Kernel 4: MLP bf16 GEMM + bias + ReLU. One wave per 16x16 output tile.
// ---------------------------------------------------------------------------
template<int KPAD, int N>
__global__ void mlp_gemm(const __bf16* __restrict__ A,
                         const __bf16* __restrict__ Wfrag,
                         const float* __restrict__ bias,
                         __bf16* __restrict__ outp)
{
  const int lane = threadIdx.x;
  const int mt = blockIdx.x, nt = blockIdx.y;
  const int m  = mt * 16 + (lane & 15);
  const int ka = (lane < 16) ? 0 : 8;
  const __bf16* arow = A + (size_t)m * KPAD + ka;
  const __bf16* bfr  = Wfrag + ((size_t)nt * 32 + lane) * 16;

  constexpr int KT = KPAD / 32;
  constexpr int FRAGSTRIDE = (N / 16) * 512;
  v8f acc0 = {0.f, 0.f, 0.f, 0.f, 0.f, 0.f, 0.f, 0.f};
  v8f acc1 = acc0;
  #pragma unroll 2
  for (int kt = 0; kt + 1 < KT; kt += 2) {
    const v16bf a0 = load_afrag(arow + (kt    ) * 32);
    const v16bf b0 = *(const v16bf*)(bfr + (size_t)(kt    ) * FRAGSTRIDE);
    const v16bf a1 = load_afrag(arow + (kt + 1) * 32);
    const v16bf b1 = *(const v16bf*)(bfr + (size_t)(kt + 1) * FRAGSTRIDE);
    acc0 = __builtin_amdgcn_wmma_f32_16x16x32_bf16(false, a0, false, b0,
                                                   (short)0, acc0, false, false);
    acc1 = __builtin_amdgcn_wmma_f32_16x16x32_bf16(false, a1, false, b1,
                                                   (short)0, acc1, false, false);
  }
  if constexpr (KT & 1) {                // odd tail (K=224 layer)
    const v16bf a0 = load_afrag(arow + (KT - 1) * 32);
    const v16bf b0 = *(const v16bf*)(bfr + (size_t)(KT - 1) * FRAGSTRIDE);
    acc0 = __builtin_amdgcn_wmma_f32_16x16x32_bf16(false, a0, false, b0,
                                                   (short)0, acc0, false, false);
  }

  const int n    = nt * 16 + (lane & 15);
  const int mrow = mt * 16 + ((lane < 16) ? 0 : 8);
  const float bv = bias[n];
  #pragma unroll
  for (int v = 0; v < 8; ++v) {
    float x = acc0[v] + acc1[v] + bv;
    x = fmaxf(x, 0.f);
    outp[(size_t)(mrow + v) * N + n] = (__bf16)x;
  }
}

// ---------------------------------------------------------------------------
// Kernel 5: cin_feat reduction + both 1-wide heads + sigmoid
// ---------------------------------------------------------------------------
__global__ void final_head(const float* __restrict__ linout,
                           const float* __restrict__ res0,
                           const float* __restrict__ res1,
                           const float* __restrict__ clW,
                           const float* __restrict__ clb,
                           const __bf16* __restrict__ h3,
                           const float* __restrict__ W4,
                           const float* __restrict__ b4,
                           float* __restrict__ outp)
{
  const int b = blockIdx.x * blockDim.x + threadIdx.x;
  if (b >= BATCH) return;

  float cin = clb[0];
  for (int o = 0; o < 128; ++o) {
    float s0 = 0.f, s1 = 0.f;
    const float* p0 = res0 + ((size_t)b * 128 + o) * 8;
    const float* p1 = res1 + ((size_t)b * 128 + o) * 8;
    #pragma unroll
    for (int k = 0; k < 8; ++k) { s0 += p0[k]; s1 += p1[k]; }
    cin += s0 * clW[o] + s1 * clW[128 + o];
  }
  cin = fmaxf(cin, 0.f);

  float dn = b4[0];
  #pragma unroll 8
  for (int j = 0; j < 64; ++j) dn += (float)h3[(size_t)b * 64 + j] * W4[j];
  dn = fmaxf(dn, 0.f);

  const float z = linout[b] + cin + dn;
  outp[b] = 1.f / (1.f + __expf(-z));
}

// ---------------------------------------------------------------------------
extern "C" void kernel_launch(void* const* d_in, const int* in_sizes, int n_in,
                              void* d_out, int out_size, void* d_ws, size_t ws_size,
                              hipStream_t stream)
{
  (void)in_sizes; (void)n_in; (void)out_size; (void)ws_size;

  const float* inputs = (const float*)d_in[0];
  const float* embt   = (const float*)d_in[1];
  const float* linW   = (const float*)d_in[2];
  const float* linb   = (const float*)d_in[3];
  const float* Wc0    = (const float*)d_in[4];
  const float* bc0    = (const float*)d_in[5];
  const float* Wc1    = (const float*)d_in[6];
  const float* bc1    = (const float*)d_in[7];
  const float* clW    = (const float*)d_in[8];
  const float* clb    = (const float*)d_in[9];
  const float* W1     = (const float*)d_in[10];
  const float* b1     = (const float*)d_in[11];
  const float* W2     = (const float*)d_in[12];
  const float* b2     = (const float*)d_in[13];
  const float* W3     = (const float*)d_in[14];
  const float* b3     = (const float*)d_in[15];
  const float* W4     = (const float*)d_in[16];
  const float* b4     = (const float*)d_in[17];
  float* out = (float*)d_out;

  // ---- workspace carve (deterministic, ~44 MB) ----
  char* ws = (char*)d_ws;
  size_t off = 0;
  auto carve = [&](size_t bytes) -> char* {
    char* p = ws + off;
    off = (off + bytes + 255) & ~(size_t)255;
    return p;
  };
  float*  X0f    = (float*)  carve((size_t)BATCH * 26 * 8 * 4);
  __bf16* h0     = (__bf16*) carve((size_t)BATCH * H0W * 2);
  float*  linout = (float*)  carve((size_t)BATCH * 4);
  __bf16* wc0    = (__bf16*) carve((size_t)704  * 128 * 2);
  __bf16* wc1    = (__bf16*) carve((size_t)3328 * 128 * 2);
  __bf16* wd1    = (__bf16*) carve((size_t)224  * 256 * 2);
  __bf16* wd2    = (__bf16*) carve((size_t)256  * 128 * 2);
  __bf16* wd3    = (__bf16*) carve((size_t)128  * 64  * 2);
  float*  res0   = (float*)  carve((size_t)BATCH * 128 * 8 * 4);
  float*  res1   = (float*)  carve((size_t)BATCH * 128 * 8 * 4);
  __bf16* h1     = (__bf16*) carve((size_t)BATCH * 256 * 2);
  __bf16* h2     = (__bf16*) carve((size_t)BATCH * 128 * 2);
  __bf16* h3     = (__bf16*) carve((size_t)BATCH * 64  * 2);

  // ---- 1. gather + inputs ----
  prep_gather<<<BATCH, 256, 0, stream>>>(inputs, embt, linW, linb, X0f, h0, linout);

  // ---- 2. weight swizzles (bf16 WMMA B fragments) ----
  auto swz = [&](const float* s, __bf16* d, int Kf, int Kp, int N, int nMajor) {
    const int total = Kp * N;
    swizzle_wfrag<<<(total + 255) / 256, 256, 0, stream>>>(s, d, Kf, N, nMajor, total);
  };
  swz(Wc0, wc0,  676,  704, 128, 1);   // (o,i,j) -> K = i*26+j
  swz(Wc1, wc1, 3328, 3328, 128, 1);
  swz(W1,  wd1,  221,  224, 256, 0);
  swz(W2,  wd2,  256,  256, 128, 0);
  swz(W3,  wd3,  128,  128,  64, 0);

  // ---- 3. CIN layers (M = 32768 rows, N = 128) ----
  cin_layer<704, 676, 26><<<KROWS / 16, 256, 16 * 704 * 2, stream>>>(
      X0f, X0f, wc0, bc0, res0);
  cin_layer<3328, 3328, 128><<<KROWS / 16, 256, 16 * 3328 * 2, stream>>>(
      res0, X0f, wc1, bc1, res1);

  // ---- 4. MLP ----
  mlp_gemm<224, 256><<<dim3(BATCH / 16, 256 / 16), 32, 0, stream>>>(h0, wd1, b1, h1);
  mlp_gemm<256, 128><<<dim3(BATCH / 16, 128 / 16), 32, 0, stream>>>(h1, wd2, b2, h2);
  mlp_gemm<128,  64><<<dim3(BATCH / 16,  64 / 16), 32, 0, stream>>>(h2, wd3, b3, h3);

  // ---- 5. heads + sigmoid ----
  final_head<<<BATCH / 256, 256, 0, stream>>>(linout, res0, res1, clW, clb,
                                              h3, W4, b4, out);
}
